// StructuralLuongAttention_36661840839074
// MI455X (gfx1250) — matrix operand order repaired and checked by reading, compiled, not run on hardware
//
#include <hip/hip_runtime.h>
#include <hip/hip_bf16.h>
#include <math.h>

typedef __attribute__((ext_vector_type(16))) __bf16 v16bf;
typedef __attribute__((ext_vector_type(8)))  float  v8f;

#define B_SZ   32
#define L_SEQ  2048
#define D_V    1024
#define NU     512

// ---------------- workspace layout (element offsets) ----------------
// [0 .. 524287]  bf16 : WvT (512 x 1024), transposed Wv
// float region starts at float index 262144:
#define WS_LEN_OFF   262144            // 32 floats: input_len per batch
#define WS_BASE_OFF  262208            // 32*512 floats: fused per-batch base
#define WS_SCORE_OFF (262208 + B_SZ*NU)            // 65536 floats: scores
#define WS_PART_OFF  (WS_SCORE_OFF + B_SZ*L_SEQ)   // 8*32*1024 floats: ctx partials

// ---------------- small prep kernels ----------------

__global__ void len_kernel(const float* __restrict__ mask, float* __restrict__ len) {
    __shared__ float red[256];
    const int b = blockIdx.x, tid = threadIdx.x;
    float s = 0.f;
    for (int l = tid; l < L_SEQ; l += 256)
        s += (mask[b * L_SEQ + l] == 0.0f) ? 1.0f : 0.0f;
    red[tid] = s; __syncthreads();
    for (int off = 128; off; off >>= 1) {
        if (tid < off) red[tid] += red[tid + off];
        __syncthreads();
    }
    if (tid == 0) len[b] = 1.0f + red[0];
}

__global__ void wvt_kernel(const float* __restrict__ Wv, __bf16* __restrict__ WvT) {
    const int i = blockIdx.x * 256 + threadIdx.x;   // i = u*1024 + k
    const int u = i >> 10, k = i & 1023;
    WvT[i] = (__bf16)Wv[k * NU + u];
}

// base[b][u] = query[b]@Wh[:,u] + bh+bv+bp+bm + log(1+t)*Wp0[u] + log(len_b)*Wp2[u]
__global__ void base_kernel(const float* __restrict__ query, const float* __restrict__ Wh,
                            const float* __restrict__ bh, const float* __restrict__ bv,
                            const float* __restrict__ Wp, const float* __restrict__ bp,
                            const float* __restrict__ bm, const int* __restrict__ timestep,
                            const float* __restrict__ len, float* __restrict__ base) {
    const int u = blockIdx.y * 256 + threadIdx.x;
    const int b = blockIdx.x;
    float acc = 0.f;
    #pragma unroll 8
    for (int k = 0; k < 1024; k++)
        acc = fmaf(query[b * 1024 + k], Wh[k * NU + u], acc);
    const float lt = __logf(1.0f + (float)timestep[0]);
    const float ll = __logf(len[b]);
    base[b * NU + u] = acc + bh[u] + bv[u] + bp[u] + bm[u]
                     + lt * Wp[0 * NU + u] + ll * Wp[2 * NU + u];
}

// ---------------- main WMMA score kernel ----------------
// block: 64 rows of flattened (b,l) x all 512 cols. 8 waves, each wave: 64 cols.
__global__ __launch_bounds__(256)
void score_wmma_kernel(const float* __restrict__ values,
                       const float* __restrict__ prev,    // previous_scores (B,L)
                       const float* __restrict__ mask,    // (B,L)
                       const float* __restrict__ Wp,      // (3,512)
                       const float* __restrict__ Wm,      // (5,512)
                       const float* __restrict__ vw,      // (512)
                       const float* __restrict__ vb,      // (1)
                       const __bf16* __restrict__ WvT,    // (512,1024) bf16
                       const float* __restrict__ base,    // (32,512)
                       float* __restrict__ scores) {      // (B*L)
    const int tid  = threadIdx.x;
    const int wave = tid >> 5;
    const int lane = tid & 31;
    const int b    = blockIdx.x >> 5;          // 32 blocks per batch
    const int l0   = (blockIdx.x & 31) << 6;   // 64 rows per block

    __shared__ __bf16 Asmem[64][136];          // 64 rows x 128 K (bf16), padded
    __shared__ float  prevS[68];               // prev window l0-2 .. l0+65
    __shared__ float  partial[8][64];

    if (tid < 68) {
        const int l = l0 - 2 + tid;
        prevS[tid] = (l >= 0 && l < L_SEQ) ? prev[b * L_SEQ + l] : 0.0f;
    }

    v8f acc[4][4];
    #pragma unroll
    for (int rt = 0; rt < 4; rt++)
        #pragma unroll
        for (int c = 0; c < 4; c++)
            #pragma unroll
            for (int i = 0; i < 8; i++) acc[rt][c][i] = 0.0f;

    const int arow  = lane & 15;
    const int halfA = (lane >> 4) << 4;        // A: K offset 0 / 16
    const int halfB = (lane >> 4) << 4;        // B: K offset 0 / 16

    for (int kc = 0; kc < D_V; kc += 128) {
        __syncthreads();
        // stage 64 rows x 128 floats -> bf16 LDS (256 thr x 8 float4)
        #pragma unroll
        for (int rep = 0; rep < 8; rep++) {
            const int idx = tid + rep * 256;       // 2048 float4 total
            const int r   = idx >> 5;              // 32 float4 per row
            const int c4  = (idx & 31) << 2;
            const float4 v = *(const float4*)(values +
                ((size_t)(b * L_SEQ + l0 + r)) * D_V + kc + c4);
            union { __bf16 h[4]; float2 f2; } pk;
            pk.h[0] = (__bf16)v.x; pk.h[1] = (__bf16)v.y;
            pk.h[2] = (__bf16)v.z; pk.h[3] = (__bf16)v.w;
            *(float2*)&Asmem[r][c4] = pk.f2;
        }
        __syncthreads();

        #pragma unroll
        for (int ki = 0; ki < 128; ki += 32) {
            // B fragments: 4 col tiles, reused across 4 row tiles
            union { v16bf v; float4 f2[2]; } bf[4];
            #pragma unroll
            for (int c = 0; c < 4; c++) {
                const int u = wave * 64 + c * 16 + (lane & 15);
                const __bf16* bp = WvT + (size_t)u * D_V + kc + ki + halfB;
                bf[c].f2[0] = *(const float4*)(bp);
                bf[c].f2[1] = *(const float4*)(bp + 8);
            }
            #pragma unroll
            for (int rt = 0; rt < 4; rt++) {
                union { v16bf v; float4 f2[2]; } a;
                const __bf16* ap = &Asmem[rt * 16 + arow][ki + halfA];
                a.f2[0] = *(const float4*)(ap);
                a.f2[1] = *(const float4*)(ap + 8);
                #pragma unroll
                for (int c = 0; c < 4; c++)
                    acc[rt][c] = __builtin_amdgcn_wmma_f32_16x16x32_bf16(
                        false, a.v, false, bf[c].v, (short)0, acc[rt][c],
                        false, false);
            }
        }
    }

    // ---- epilogue: fused adds, tanh, dot with vw, reduce over 512 cols ----
    const int col15 = lane & 15;
    const int r0    = (lane >> 4) << 3;        // C rows: r0 + 0..7
    float rsum[4][8];
    #pragma unroll
    for (int rt = 0; rt < 4; rt++)
        #pragma unroll
        for (int r = 0; r < 8; r++) rsum[rt][r] = 0.f;

    #pragma unroll
    for (int c = 0; c < 4; c++) {
        const int u   = wave * 64 + c * 16 + col15;
        const float bs  = base[b * NU + u];
        const float wp1 = Wp[1 * NU + u];
        const float vwu = vw[u];
        float wm[5];
        #pragma unroll
        for (int j = 0; j < 5; j++) wm[j] = Wm[j * NU + u];
        #pragma unroll
        for (int rt = 0; rt < 4; rt++) {
            #pragma unroll
            for (int r = 0; r < 8; r++) {
                const int rr = rt * 16 + r0 + r;          // row within block
                float mar = 0.f;
                #pragma unroll
                for (int j = 0; j < 5; j++) mar = fmaf(prevS[rr + j], wm[j], mar);
                const float val = acc[rt][c][r] + bs
                                + __logf(2.0f + (float)(l0 + rr)) * wp1 + mar;
                rsum[rt][r] += tanhf(val) * vwu;
            }
        }
    }
    // butterfly within 16-lane halves (cols of a tile)
    #pragma unroll
    for (int m = 1; m < 16; m <<= 1)
        #pragma unroll
        for (int rt = 0; rt < 4; rt++)
            #pragma unroll
            for (int r = 0; r < 8; r++)
                rsum[rt][r] += __shfl_xor(rsum[rt][r], m, 32);

    if ((lane & 15) == 0) {
        #pragma unroll
        for (int rt = 0; rt < 4; rt++)
            #pragma unroll
            for (int r = 0; r < 8; r++)
                partial[wave][rt * 16 + r0 + r] = rsum[rt][r];
    }
    __syncthreads();
    if (tid < 64) {
        float s = 0.f;
        #pragma unroll
        for (int w = 0; w < 8; w++) s += partial[w][tid];
        const int l = l0 + tid;
        s += vb[0];
        s += mask[b * L_SEQ + l] * -1000000000.0f;
        scores[b * L_SEQ + l] = s;
    }
}

// ---------------- softmax over L per batch ----------------
__global__ void softmax_kernel(const float* __restrict__ scores, float* __restrict__ weights) {
    __shared__ float red[256];
    const int b = blockIdx.x, tid = threadIdx.x;
    float m = -INFINITY;
    for (int l = tid; l < L_SEQ; l += 256) m = fmaxf(m, scores[b * L_SEQ + l]);
    red[tid] = m; __syncthreads();
    for (int off = 128; off; off >>= 1) {
        if (tid < off) red[tid] = fmaxf(red[tid], red[tid + off]);
        __syncthreads();
    }
    m = red[0]; __syncthreads();
    float s = 0.f;
    for (int l = tid; l < L_SEQ; l += 256) s += __expf(scores[b * L_SEQ + l] - m);
    red[tid] = s; __syncthreads();
    for (int off = 128; off; off >>= 1) {
        if (tid < off) red[tid] += red[tid + off];
        __syncthreads();
    }
    const float inv = 1.0f / red[0];
    for (int l = tid; l < L_SEQ; l += 256)
        weights[b * L_SEQ + l] = __expf(scores[b * L_SEQ + l] - m) * inv;
}

// ---------------- context = sum_l w[l]*values[b,l,:] (2-stage, no atomics) --
__global__ void ctx_part_kernel(const float* __restrict__ values,
                                const float* __restrict__ weights,
                                float* __restrict__ part) {
    const int b = blockIdx.x, dchunk = blockIdx.y, lchunk = blockIdx.z;
    const int d = dchunk * 256 + threadIdx.x;
    const int lbase = lchunk * 256;
    float acc = 0.f;
    for (int l = 0; l < 256; l++) {
        const float w = weights[b * L_SEQ + lbase + l];
        acc = fmaf(w, values[((size_t)(b * L_SEQ + lbase + l)) * D_V + d], acc);
    }
    part[((size_t)lchunk * B_SZ + b) * D_V + d] = acc;
}

__global__ void ctx_reduce_kernel(const float* __restrict__ part, float* __restrict__ ctx) {
    const int i = blockIdx.x * 256 + threadIdx.x;   // b*1024 + d, 32768 total
    float s = 0.f;
    #pragma unroll
    for (int c = 0; c < 8; c++) s += part[(size_t)c * (B_SZ * D_V) + i];
    ctx[i] = s;
}

// ---------------- launcher ----------------
extern "C" void kernel_launch(void* const* d_in, const int* in_sizes, int n_in,
                              void* d_out, int out_size, void* d_ws, size_t ws_size,
                              hipStream_t stream) {
    const float* query  = (const float*)d_in[0];
    const float* values = (const float*)d_in[1];
    const float* mask   = (const float*)d_in[2];
    const float* prev   = (const float*)d_in[3];
    const int*   tstep  = (const int*)  d_in[4];
    const float* Wh     = (const float*)d_in[5];
    const float* bh     = (const float*)d_in[6];
    const float* Wv     = (const float*)d_in[7];
    const float* bv     = (const float*)d_in[8];
    const float* Wp     = (const float*)d_in[9];
    const float* bp     = (const float*)d_in[10];
    const float* bm     = (const float*)d_in[11];
    const float* bm2    = (const float*)d_in[12];
    const float* vw     = (const float*)d_in[13];
    const float* vb     = (const float*)d_in[14];
    // note: d_in[11] = Wm (5,512), d_in[12] = bm (512)
    const float* Wm = bm;     // (renaming for clarity below)
    const float* bm_vec = bm2;

    float*  wsf    = (float*)d_ws;
    __bf16* WvT    = (__bf16*)d_ws;
    float*  len    = wsf + WS_LEN_OFF;
    float*  base   = wsf + WS_BASE_OFF;
    float*  scores = wsf + WS_SCORE_OFF;
    float*  part   = wsf + WS_PART_OFF;

    float* ctx     = (float*)d_out;                 // (32,1024)
    float* weights = (float*)d_out + B_SZ * D_V;    // (32,2048)

    len_kernel<<<B_SZ, 256, 0, stream>>>(mask, len);
    wvt_kernel<<<(NU * D_V) / 256, 256, 0, stream>>>(Wv, WvT);
    base_kernel<<<dim3(B_SZ, NU / 256), 256, 0, stream>>>(
        query, Wh, bh, bv, Wp, bp, bm_vec, tstep, len, base);
    score_wmma_kernel<<<(B_SZ * L_SEQ) / 64, 256, 0, stream>>>(
        values, prev, mask, Wp, Wm, vw, vb, WvT, base, scores);
    softmax_kernel<<<B_SZ, 256, 0, stream>>>(scores, weights);
    ctx_part_kernel<<<dim3(B_SZ, D_V / 256, L_SEQ / 256), 256, 0, stream>>>(
        values, weights, part);
    ctx_reduce_kernel<<<(B_SZ * D_V) / 256, 256, 0, stream>>>(part, ctx);
}
